// SelfAttention_50989851738367
// MI455X (gfx1250) — matrix operand order repaired and checked by reading, not compiled, not run on hardware
//
#include <hip/hip_runtime.h>
#include <hip/hip_bf16.h>
#include <cstdint>

#define D_EMBED   1024
#define NUM_HEADS 16
#define D_HEAD    64
#define BATCH     2
#define SEQ       2048
#define ROWS      (BATCH * SEQ)     // 4096 token rows

typedef __attribute__((ext_vector_type(16))) __bf16 v16bf;
typedef __attribute__((ext_vector_type(8)))  float  v8f;

union FragU { uint4 q[2]; v16bf v; };

__device__ __forceinline__ unsigned short f32_to_bf16(float f) {
    uint32_t u = __builtin_bit_cast(uint32_t, f);
    u += 0x7FFFu + ((u >> 16) & 1u);          // round-to-nearest-even
    return (unsigned short)(u >> 16);
}

// ---- WMMA fragment loaders (bf16, 16x16x32 shapes) -------------------------
// A (16x32): lane holds row m = l%16.
//   lanes 0-15 : v0-3 = K 0..7,  v4-7 = K 16..23
//   lanes16-31 : v0-3 = K 8..15, v4-7 = K 24..31
__device__ __forceinline__ v16bf load_a(const unsigned short* p, int ld,
                                        int m0, int k0, int lane) {
    const unsigned short* r =
        p + (size_t)(m0 + (lane & 15)) * ld + k0 + ((lane >> 4) << 3);
    FragU f;
    f.q[0] = *(const uint4*)(r);        // 8 bf16 = 16B
    f.q[1] = *(const uint4*)(r + 16);   // 8 bf16 = 16B
    return f.v;
}

// B (32x16) sourced from a row-major matrix whose ROWS are B's columns
// (i.e. B = M^T): lane holds column n = l%16; lanes0-15: K 0..7, 16..23,
// lanes16-31: K 8..15, 24..31, packed 2 values per VGPR, K contiguous in memory.
__device__ __forceinline__ v16bf load_b(const unsigned short* p, int ld,
                                        int n0, int k0, int lane) {
    const unsigned short* r =
        p + (size_t)(n0 + (lane & 15)) * ld + k0 + ((lane >> 4) << 3);
    FragU f;
    f.q[0] = *(const uint4*)(r);        // K 8h..8h+7
    f.q[1] = *(const uint4*)(r + 16);   // K 16+8h..23+8h
    return f.v;
}

// ---- fp32 -> bf16 convert --------------------------------------------------
__global__ void cvt_f32_bf16(const float* __restrict__ src,
                             unsigned short* __restrict__ dst, int n) {
    int i = blockIdx.x * blockDim.x + threadIdx.x;
    if (i < n) dst[i] = f32_to_bf16(src[i]);
}

// ---- fused QKV projection: out = X @ W^T + b, reshaped per head ------------
// grid: (ROWS/16, D_EMBED/64), block 128 (4 waves, one 16x16 tile each).
// transposed==0: dst[b][h][s][d]   (Q, K)
// transposed==1: dst[b][h][d][s]   (V^T, so ctx B-fragments are contiguous)
__global__ __launch_bounds__(128)
void qkv_proj(const unsigned short* __restrict__ Xb,
              const unsigned short* __restrict__ Wb,
              const float* __restrict__ bias,
              unsigned short* __restrict__ dst, int transposed) {
    const int lane = threadIdx.x & 31, wave = threadIdx.x >> 5;
    const int s0 = blockIdx.x * 16;
    const int e0 = blockIdx.y * 64 + wave * 16;
    const int n  = e0 + (lane & 15);

    float bv = bias[n];                         // C-layout: every VGPR same N
    v8f c = { bv, bv, bv, bv, bv, bv, bv, bv };

    #pragma unroll 4
    for (int k0 = 0; k0 < D_EMBED; k0 += 32) {
        v16bf a = load_a(Xb, D_EMBED, s0, k0, lane);   // X row-major: A direct
        v16bf b = load_b(Wb, D_EMBED, e0, k0, lane);   // B = W^T: W rows = B cols
        c = __builtin_amdgcn_wmma_f32_16x16x32_bf16(false, a, false, b,
                                                    (short)0, c, false, false);
    }

    const int mb = (lane >> 4) << 3;            // M = j (+8 for hi lanes)
    const int h = n >> 6, d = n & 63;
    #pragma unroll
    for (int j = 0; j < 8; ++j) {
        int s  = s0 + j + mb;
        int bi = s >> 11, si = s & (SEQ - 1);
        size_t idx = transposed
            ? ((size_t)(bi * NUM_HEADS + h) * D_HEAD + d) * SEQ + si
            : ((size_t)(bi * NUM_HEADS + h) * SEQ + si) * D_HEAD + d;
        dst[idx] = f32_to_bf16(c[j]);
    }
}

// ---- fused attention: scores -> softmax (LDS) -> probs out + ctx WMMA ------
// grid: (SEQ/16, BATCH*NUM_HEADS), block 128, dyn-LDS 192 KB:
//   sc[16][2048] fp32 (128 KB)  +  pb[16][2048] bf16 (64 KB)
#define SMEM_BYTES (16 * SEQ * 4 + 16 * SEQ * 2)

__global__ __launch_bounds__(128)
void attn_fused(const unsigned short* __restrict__ Qb,
                const unsigned short* __restrict__ Kb,
                const unsigned short* __restrict__ Vtb,
                float* __restrict__ ctx_out,
                float* __restrict__ probs_out) {
    extern __shared__ char smem[];
    float*          sc = (float*)smem;                        // [16][SEQ]
    unsigned short* pb = (unsigned short*)(smem + 16 * SEQ * 4);

    const int lane = threadIdx.x & 31, wave = threadIdx.x >> 5;
    const int q0 = blockIdx.x * 16;
    const int bh = blockIdx.y;
    const unsigned short* Qh = Qb  + (size_t)bh * SEQ * D_HEAD;
    const unsigned short* Kh = Kb  + (size_t)bh * SEQ * D_HEAD;
    const unsigned short* Vh = Vtb + (size_t)bh * D_HEAD * SEQ;   // [dh][S]

    // Phase 1: scores stripe (16 x 2048) into LDS. K rows are L2-resident.
    for (int t = wave; t < SEQ / 16; t += 4) {
        v8f c = { 0.f, 0.f, 0.f, 0.f, 0.f, 0.f, 0.f, 0.f };
        #pragma unroll
        for (int d0 = 0; d0 < D_HEAD; d0 += 32) {
            v16bf a = load_a(Qh, D_HEAD, q0, d0, lane);
            v16bf b = load_b(Kh, D_HEAD, t * 16, d0, lane);   // B = K^T
            c = __builtin_amdgcn_wmma_f32_16x16x32_bf16(false, a, false, b,
                                                        (short)0, c, false, false);
        }
        if (t + 4 < SEQ / 16)   // pull next K tile toward the WGP
            __builtin_prefetch(Kh + (size_t)((t + 4) * 16 + (lane & 15)) * D_HEAD, 0, 1);
        const int mb = (lane >> 4) << 3, nn = t * 16 + (lane & 15);
        #pragma unroll
        for (int j = 0; j < 8; ++j)
            sc[(size_t)(j + mb) * SEQ + nn] = c[j] * 0.125f;  // 1/sqrt(64)
    }
    __syncthreads();

    // Phase 2: softmax over each 2048-wide row; 8 lanes per row.
    {
        const int r = threadIdx.x >> 3, g = threadIdx.x & 7;
        float* row = sc + (size_t)r * SEQ;
        float mx = -3.402823466e38f;
        for (int k = g; k < SEQ; k += 8) mx = fmaxf(mx, row[k]);
        mx = fmaxf(mx, __shfl_xor(mx, 1));
        mx = fmaxf(mx, __shfl_xor(mx, 2));
        mx = fmaxf(mx, __shfl_xor(mx, 4));
        float sum = 0.f;
        for (int k = g; k < SEQ; k += 8) {
            float e = __expf(row[k] - mx);
            row[k] = e; sum += e;
        }
        sum += __shfl_xor(sum, 1);
        sum += __shfl_xor(sum, 2);
        sum += __shfl_xor(sum, 4);
        float inv = 1.0f / sum;
        for (int k = g; k < SEQ; k += 8) {
            float pv = row[k] * inv;
            row[k] = pv;                           // fp32 probs for output
            pb[(size_t)r * SEQ + k] = f32_to_bf16(pv); // bf16 for ctx WMMA
        }
    }
    __syncthreads();

    // Phase 3: stream the fp32 probs stripe out (coalesced float4) — the
    // 537 MB write that sets this kernel's roofline.
    {
        const float4* s4 = (const float4*)sc;
        float4* d4 = (float4*)(probs_out + (size_t)bh * SEQ * SEQ + (size_t)q0 * SEQ);
        for (int i = threadIdx.x; i < 16 * SEQ / 4; i += 128) d4[i] = s4[i];
    }

    // Phase 4: ctx = probs @ V. Wave w owns dh columns [16w, 16w+16).
    {
        const int n0 = wave * 16;
        v8f c = { 0.f, 0.f, 0.f, 0.f, 0.f, 0.f, 0.f, 0.f };
        for (int k0 = 0; k0 < SEQ; k0 += 32) {
            v16bf a = load_a(pb, SEQ, 0, k0, lane);       // probs bf16 from LDS
            v16bf b = load_b(Vh, SEQ, n0, k0, lane);      // V^T rows = B cols
            c = __builtin_amdgcn_wmma_f32_16x16x32_bf16(false, a, false, b,
                                                        (short)0, c, false, false);
        }
        const int mb = (lane >> 4) << 3;
        const int bi = bh >> 4, h = bh & 15;
        const int e = h * D_HEAD + n0 + (lane & 15);
        #pragma unroll
        for (int j = 0; j < 8; ++j) {
            int s = q0 + j + mb;
            ctx_out[(size_t)(bi * SEQ + s) * D_EMBED + e] = c[j];
        }
    }
}

// ---------------------------------------------------------------------------
extern "C" void kernel_launch(void* const* d_in, const int* in_sizes, int n_in,
                              void* d_out, int out_size, void* d_ws, size_t ws_size,
                              hipStream_t stream) {
    const float* X  = (const float*)d_in[0];
    const float* Wq = (const float*)d_in[1];
    const float* bq = (const float*)d_in[2];
    const float* Wk = (const float*)d_in[3];
    const float* bk = (const float*)d_in[4];
    const float* Wv = (const float*)d_in[5];
    const float* bv = (const float*)d_in[6];

    float* ctx   = (float*)d_out;                                  // [2,2048,1024]
    float* probs = (float*)d_out + (size_t)BATCH * SEQ * D_EMBED;  // [2,16,2048,2048]

    // Workspace carve-up (bf16 staging, ~40 MB total)
    char* ws = (char*)d_ws;
    unsigned short* Xb  = (unsigned short*)ws; ws += (size_t)ROWS * D_EMBED * 2;
    unsigned short* Wqb = (unsigned short*)ws; ws += (size_t)D_EMBED * D_EMBED * 2;
    unsigned short* Wkb = (unsigned short*)ws; ws += (size_t)D_EMBED * D_EMBED * 2;
    unsigned short* Wvb = (unsigned short*)ws; ws += (size_t)D_EMBED * D_EMBED * 2;
    unsigned short* Qb  = (unsigned short*)ws; ws += (size_t)ROWS * D_EMBED * 2;
    unsigned short* Kb  = (unsigned short*)ws; ws += (size_t)ROWS * D_EMBED * 2;
    unsigned short* Vtb = (unsigned short*)ws; ws += (size_t)ROWS * D_EMBED * 2;

    const int nX = ROWS * D_EMBED, nW = D_EMBED * D_EMBED;
    cvt_f32_bf16<<<(nX + 255) / 256, 256, 0, stream>>>(X,  Xb,  nX);
    cvt_f32_bf16<<<(nW + 255) / 256, 256, 0, stream>>>(Wq, Wqb, nW);
    cvt_f32_bf16<<<(nW + 255) / 256, 256, 0, stream>>>(Wk, Wkb, nW);
    cvt_f32_bf16<<<(nW + 255) / 256, 256, 0, stream>>>(Wv, Wvb, nW);

    dim3 pg(ROWS / 16, D_EMBED / 64);
    qkv_proj<<<pg, 128, 0, stream>>>(Xb, Wqb, bq, Qb,  0);
    qkv_proj<<<pg, 128, 0, stream>>>(Xb, Wkb, bk, Kb,  0);
    qkv_proj<<<pg, 128, 0, stream>>>(Xb, Wvb, bv, Vtb, 1);

    attn_fused<<<dim3(SEQ / 16, BATCH * NUM_HEADS), 128, SMEM_BYTES, stream>>>(
        Qb, Kb, Vtb, ctx, probs);
}
